// Attention_8701603742210
// MI455X (gfx1250) — matrix operand order repaired and checked by reading, not compiled, not run on hardware
//
#include <hip/hip_runtime.h>

#define BATCH 16
#define SEQ   1024
#define EDIM  640
#define HEADS 10
#define HDIM  64
#define QKVD  1920            // 3*EDIM
#define SCALE 0.0395284708f   // 640^-0.5 (reference uses DIM^-0.5, not head_dim)

typedef __attribute__((ext_vector_type(16))) __bf16 v16bf;
typedef __attribute__((ext_vector_type(8)))  float  v8f;

union FragA { v16bf v; unsigned u[8]; };
union FragB { v16bf v; uint4 q[2]; };
union Acc   { v8f   v; float f[8]; };

__device__ __forceinline__ unsigned short f2bf(float f) {
  unsigned u = __builtin_bit_cast(unsigned, f);
  u += 0x7FFFu + ((u >> 16) & 1u);           // round-to-nearest-even
  return (unsigned short)(u >> 16);
}

// ---- A fragment (16x32 bf16): lane holds row (lane&15), interleaved K pattern
//      K dwords: {0,2,4,6,16,18,20,22} + (lane>>4)*8
__device__ __forceinline__ v16bf load_a_global(const unsigned short* base,
                                               int row0, int col0, int ld) {
  int lane = threadIdx.x & 31;
  const unsigned short* p =
      base + (size_t)(row0 + (lane & 15)) * ld + col0 + ((lane >> 4) << 3);
  FragA f;
#pragma unroll
  for (int i = 0; i < 4; ++i) f.u[i]     = *(const unsigned*)(p + 2 * i);
#pragma unroll
  for (int i = 0; i < 4; ++i) f.u[4 + i] = *(const unsigned*)(p + 16 + 2 * i);
  return f.v;
}

__device__ __forceinline__ v16bf load_a_lds(const unsigned short* base, int ld) {
  int lane = threadIdx.x & 31;
  const unsigned short* p = base + (lane & 15) * ld + ((lane >> 4) << 3);
  FragA f;
#pragma unroll
  for (int i = 0; i < 4; ++i) f.u[i]     = *(const unsigned*)(p + 2 * i);
#pragma unroll
  for (int i = 0; i < 4; ++i) f.u[4 + i] = *(const unsigned*)(p + 16 + 2 * i);
  return f.v;
}

// ---- B fragment (32x16 bf16): lane holds column (lane&15), K contiguous
//      K = (lane>>4)*8 + {0..7, 16..23}  -> two 16B loads from a row-major-over-N matrix
__device__ __forceinline__ v16bf load_b_global(const unsigned short* base,
                                               int n0, int k0, int ld) {
  int lane = threadIdx.x & 31;
  const unsigned short* p =
      base + (size_t)(n0 + (lane & 15)) * ld + k0 + ((lane >> 4) << 3);
  FragB f;
  f.q[0] = *(const uint4*)(p);
  f.q[1] = *(const uint4*)(p + 16);
  return f.v;
}

__device__ __forceinline__ v8f wmma_bf16(v16bf a, v16bf b, v8f c) {
  return __builtin_amdgcn_wmma_f32_16x16x32_bf16(false, a, false, b,
                                                 (short)0, c, false, false);
}

// ---------------------------------------------------------------- elementwise
__global__ void convert_f32_bf16(const float* __restrict__ in,
                                 unsigned short* __restrict__ out, int n) {
  int i = blockIdx.x * blockDim.x + threadIdx.x;
  if (i < n) out[i] = f2bf(in[i]);
}

// in: fp32 [R][C]  ->  out: bf16 [C][R]
__global__ void transpose_f32_bf16(const float* __restrict__ in,
                                   unsigned short* __restrict__ out,
                                   int R, int C) {
  int i = blockIdx.x * blockDim.x + threadIdx.x;
  if (i < R * C) {
    int r = i % R;
    int c = i / R;
    out[(size_t)c * R + r] = f2bf(in[(size_t)r * C + c]);
  }
}

// gather V head-slices out of qkv into vT[b][h][d][n]  (bf16 -> bf16 copy)
__global__ void build_vT(const unsigned short* __restrict__ qkv,
                         unsigned short* __restrict__ vT, int total) {
  int i = blockIdx.x * blockDim.x + threadIdx.x;
  if (i < total) {
    int n = i % SEQ;
    int d = (i / SEQ) % HDIM;
    int h = (i / (SEQ * HDIM)) % HEADS;
    int b = i / (SEQ * HDIM * HEADS);
    vT[i] = qkv[((size_t)(b * SEQ + n)) * QKVD + 2 * EDIM + h * HDIM + d];
  }
}

// ------------------------------------------------------------------ WMMA GEMM
// C[m][n] = sum_k A[m][k] * Bt[n][k] + bias[n]
// 128 threads = 4 waves; wave owns a 32x64 C tile (2x4 accum frags).
__global__ void __launch_bounds__(128)
gemm_bf16_kernel(const unsigned short* __restrict__ A,
                 const unsigned short* __restrict__ Bt,
                 const float* __restrict__ bias,
                 unsigned short* __restrict__ outB,
                 float* __restrict__ outF,
                 int M, int Nn, int K, int lda, int ldb, int ldc) {
  int wave = threadIdx.x >> 5;
  int lane = threadIdx.x & 31;
  int m0 = blockIdx.x * 128 + wave * 32;
  int n0 = blockIdx.y * 64;

  Acc acc[2][4];
#pragma unroll
  for (int i = 0; i < 2; ++i)
#pragma unroll
    for (int j = 0; j < 4; ++j)
#pragma unroll
      for (int r = 0; r < 8; ++r) acc[i][j].f[r] = 0.0f;

  for (int k = 0; k < K; k += 32) {
    if (k + 32 < K) {  // hint next tiles into cache (global_prefetch_b8)
      __builtin_prefetch(A + (size_t)m0 * lda + k + 32, 0, 1);
      __builtin_prefetch(Bt + (size_t)n0 * ldb + k + 32, 0, 1);
    }
    v16bf a0 = load_a_global(A, m0, k, lda);
    v16bf a1 = load_a_global(A, m0 + 16, k, lda);
    v16bf b[4];
#pragma unroll
    for (int j = 0; j < 4; ++j) b[j] = load_b_global(Bt, n0 + j * 16, k, ldb);
#pragma unroll
    for (int j = 0; j < 4; ++j) {
      acc[0][j].v = wmma_bf16(a0, b[j], acc[0][j].v);
      acc[1][j].v = wmma_bf16(a1, b[j], acc[1][j].v);
    }
  }

  int colLane = lane & 15;
  int rowHalf = (lane >> 4) * 8;
#pragma unroll
  for (int i = 0; i < 2; ++i)
#pragma unroll
    for (int j = 0; j < 4; ++j) {
      int col = n0 + j * 16 + colLane;
      float bv = bias[col];
#pragma unroll
      for (int r = 0; r < 8; ++r) {
        int row = m0 + i * 16 + rowHalf + r;
        float val = acc[i][j].f[r] + bv;
        if (outF) outF[(size_t)row * ldc + col] = val;
        else      outB[(size_t)row * ldc + col] = f2bf(val);
      }
    }
}

// -------------------------------------------------------- flash attention
// 1 wave per (b, h, 16-row q tile). Online softmax over j-tiles of 32.
__global__ void __launch_bounds__(32)
attn_kernel(const unsigned short* __restrict__ qkv,
            const unsigned short* __restrict__ vT,
            unsigned short* __restrict__ attnout) {
  __shared__ unsigned short Plds[16 * 32];

  int lane = threadIdx.x & 31;
  int q0 = blockIdx.x * 16;
  int h  = blockIdx.y;
  int b  = blockIdx.z;

  const unsigned short* qbase = qkv + (size_t)b * SEQ * QKVD + h * HDIM;
  const unsigned short* kbase = qbase + EDIM;
  const unsigned short* vbase = vT + ((size_t)(b * HEADS + h) * HDIM) * SEQ;

  // Q rows q0..q0+15, d=0..63 -> two A fragments, loaded once
  v16bf aq0 = load_a_global(qbase, q0, 0, QKVD);
  v16bf aq1 = load_a_global(qbase, q0, 32, QKVD);

  Acc o[4];
  float mrow[8], lrow[8];
#pragma unroll
  for (int r = 0; r < 8; ++r) { mrow[r] = -3.0e38f; lrow[r] = 0.0f; }
#pragma unroll
  for (int db = 0; db < 4; ++db)
#pragma unroll
    for (int r = 0; r < 8; ++r) o[db].f[r] = 0.0f;

  for (int jt = 0; jt < SEQ; jt += 32) {
    // S tiles: cols jt..+15 (s0) and jt+16..+31 (s1)
    Acc s0, s1;
#pragma unroll
    for (int r = 0; r < 8; ++r) { s0.f[r] = 0.0f; s1.f[r] = 0.0f; }

    v16bf k0a = load_b_global(kbase, jt,      0,  QKVD);
    v16bf k0b = load_b_global(kbase, jt,      32, QKVD);
    s0.v = wmma_bf16(aq0, k0a, s0.v);
    s0.v = wmma_bf16(aq1, k0b, s0.v);
    v16bf k1a = load_b_global(kbase, jt + 16, 0,  QKVD);
    v16bf k1b = load_b_global(kbase, jt + 16, 32, QKVD);
    s1.v = wmma_bf16(aq0, k1a, s1.v);
    s1.v = wmma_bf16(aq1, k1b, s1.v);

    // online softmax; row reductions across the 16-lane column group
    // (xor 1/2/4/8 keeps lanes within their half: rows 0-7 vs 8-15)
#pragma unroll
    for (int r = 0; r < 8; ++r) {
      float x0 = s0.f[r] * SCALE;
      float x1 = s1.f[r] * SCALE;
      float mx = fmaxf(x0, x1);
      mx = fmaxf(mx, __shfl_xor(mx, 1, 32));
      mx = fmaxf(mx, __shfl_xor(mx, 2, 32));
      mx = fmaxf(mx, __shfl_xor(mx, 4, 32));
      mx = fmaxf(mx, __shfl_xor(mx, 8, 32));
      float newm = fmaxf(mrow[r], mx);
      float alpha = __expf(mrow[r] - newm);
      float p0 = __expf(x0 - newm);
      float p1 = __expf(x1 - newm);
      float ps = p0 + p1;
      ps += __shfl_xor(ps, 1, 32);
      ps += __shfl_xor(ps, 2, 32);
      ps += __shfl_xor(ps, 4, 32);
      ps += __shfl_xor(ps, 8, 32);
      lrow[r] = lrow[r] * alpha + ps;
      mrow[r] = newm;
#pragma unroll
      for (int db = 0; db < 4; ++db) o[db].f[r] *= alpha;
      s0.f[r] = p0;
      s1.f[r] = p1;
    }

    // repack P: f32 C-layout -> bf16 A-layout via LDS
    {
      int col = lane & 15;
      int rbase = (lane >> 4) * 8;
#pragma unroll
      for (int r = 0; r < 8; ++r) {
        Plds[(rbase + r) * 32 + col]      = f2bf(s0.f[r]);
        Plds[(rbase + r) * 32 + col + 16] = f2bf(s1.f[r]);
      }
    }
    __syncthreads();
    v16bf pa = load_a_lds(Plds, 32);
#pragma unroll
    for (int db = 0; db < 4; ++db) {
      v16bf vb = load_b_global(vbase, db * 16, jt, SEQ);  // Vt[d][j]
      o[db].v = wmma_bf16(pa, vb, o[db].v);
    }
    __syncthreads();
  }

  // normalize and write head-slice (bf16) into attnout[b*N+row][h*64+d]
  int col = lane & 15;
  int rbase = (lane >> 4) * 8;
#pragma unroll
  for (int r = 0; r < 8; ++r) {
    float inv = 1.0f / lrow[r];
    int row = q0 + rbase + r;
    size_t outRow = ((size_t)b * SEQ + row) * EDIM + h * HDIM;
#pragma unroll
    for (int db = 0; db < 4; ++db)
      attnout[outRow + db * 16 + col] = f2bf(o[db].f[r] * inv);
  }
}

// ---------------------------------------------------------------------------
extern "C" void kernel_launch(void* const* d_in, const int* in_sizes, int n_in,
                              void* d_out, int out_size, void* d_ws, size_t ws_size,
                              hipStream_t stream) {
  const float* x     = (const float*)d_in[0];
  const float* w_qkv = (const float*)d_in[1];
  const float* b_qkv = (const float*)d_in[2];
  const float* w_out = (const float*)d_in[3];
  const float* b_out = (const float*)d_in[4];
  float* out = (float*)d_out;

  // workspace layout (bf16 buffers), ~129 MB total
  char* ws = (char*)d_ws;
  unsigned short* xb      = (unsigned short*)(ws);                 // 16384x640
  unsigned short* wqkvT   = (unsigned short*)(ws + 20971520);      // 1920x640
  unsigned short* woutT   = (unsigned short*)(ws + 23429120);      // 640x640
  unsigned short* qkv     = (unsigned short*)(ws + 24248320);      // 16384x1920
  unsigned short* vT      = (unsigned short*)(ws + 87162880);      // 16x10x64x1024
  unsigned short* attnout = (unsigned short*)(ws + 108134400);     // 16384x640

  const int M = BATCH * SEQ;  // 16384

  int nx = M * EDIM;
  convert_f32_bf16<<<(nx + 255) / 256, 256, 0, stream>>>(x, xb, nx);

  int nwq = EDIM * QKVD;
  transpose_f32_bf16<<<(nwq + 255) / 256, 256, 0, stream>>>(w_qkv, wqkvT, EDIM, QKVD);
  int nwo = EDIM * EDIM;
  transpose_f32_bf16<<<(nwo + 255) / 256, 256, 0, stream>>>(w_out, woutT, EDIM, EDIM);

  // qkv = x @ w_qkv + b_qkv  (bf16 out)
  dim3 g1(M / 128, QKVD / 64);
  gemm_bf16_kernel<<<g1, 128, 0, stream>>>(xb, wqkvT, b_qkv, qkv, nullptr,
                                           M, QKVD, EDIM, EDIM, EDIM, QKVD);

  int nv = BATCH * HEADS * HDIM * SEQ;
  build_vT<<<(nv + 255) / 256, 256, 0, stream>>>(qkv, vT, nv);

  dim3 ga(SEQ / 16, HEADS, BATCH);
  attn_kernel<<<ga, 32, 0, stream>>>(qkv, vT, attnout);

  // out = attnout @ w_out + b_out  (fp32 out)
  dim3 g2(M / 128, EDIM / 64);
  gemm_bf16_kernel<<<g2, 128, 0, stream>>>(attnout, woutT, b_out, nullptr, out,
                                           M, EDIM, EDIM, EDIM, EDIM, EDIM);
}
